// InParAttention_45372034515021
// MI455X (gfx1250) — matrix operand order repaired and checked by reading, not compiled, not run on hardware
//
#include <hip/hip_runtime.h>
#include <math.h>

#define B_    2
#define L_    256
#define S_    256
#define H_    8
#define D_    64
#define U_    27
#define NREST 229
#define NFQ   14
#define NFK   129

typedef __attribute__((ext_vector_type(16))) _Float16 v16h;
typedef __attribute__((ext_vector_type(8)))  float    v8f;

// Hardware transcendental tanh (gfx1250 v_tanh_f32). TRANS32 result needs one
// independent op before use (ISA 7.4); the compiler cannot see into asm, so we
// pad with v_nop.
__device__ __forceinline__ float htanh(float x) {
  float r;
  asm volatile("v_tanh_f32 %0, %1\n\tv_nop" : "=v"(r) : "v"(x));
  return r;
}

// ---- WMMA fragment loaders (wave32, 16x16x32 f16) -------------------------
// A tile in LDS: row-major [m][k], stride 32 halves.
__device__ __forceinline__ v16h load_fragA(const _Float16* tile) {
  const int lane = threadIdx.x & 31;
  const _Float16* p = tile + (lane & 15) * 32 + ((lane >> 4) << 3);
  v16h f;
#pragma unroll
  for (int i = 0; i < 8; ++i) { f[i] = p[i]; f[8 + i] = p[16 + i]; }
  return f;
}
// B tile in LDS stored transposed: [n][k], stride 32 halves.
__device__ __forceinline__ v16h load_fragB(const _Float16* tileT) {
  const int lane = threadIdx.x & 31;
  const _Float16* p = tileT + (lane & 15) * 32 + ((lane >> 4) << 3);
  v16h f;
#pragma unroll
  for (int i = 0; i < 8; ++i) { f[i] = p[i]; f[8 + i] = p[16 + i]; }
  return f;
}

// ---- Kernel 1: naive rfft (DFT) of keys & values along S (256 -> 129 bins) -
__global__ void dft_kv_kernel(const float* __restrict__ keys,
                              const float* __restrict__ values,
                              float* kf_re, float* kf_im,
                              float* vf_re, float* vf_im) {
  int id = blockIdx.x * blockDim.x + threadIdx.x;
  if (id >= B_ * NFK * H_ * D_) return;
  int d = id & 63;
  int h = (id >> 6) & 7;
  int rem = id >> 9;           // b*NFK + f
  int f = rem % NFK;
  int b = rem / NFK;
  float step = -6.2831853071795864f * (float)f / 256.0f;
  float kr = 0.f, ki = 0.f, vr = 0.f, vi = 0.f;
  for (int s = 0; s < S_; ++s) {
    int o = ((b * S_ + s) * H_ + h) * D_ + d;
    if (s + 2 < S_) {          // stream hint: 2KB-strided walk -> global_prefetch_b8
      __builtin_prefetch(&keys[o + 2 * H_ * D_], 0, 0);
      __builtin_prefetch(&values[o + 2 * H_ * D_], 0, 0);
    }
    float sn, cs;
    __sincosf(step * (float)s, &sn, &cs);
    float kv = keys[o], vv = values[o];
    kr += kv * cs; ki += kv * sn;
    vr += vv * cs; vi += vv * sn;
  }
  kf_re[id] = kr; kf_im[id] = ki;
  vf_re[id] = vr; vf_im[id] = vi;
}

// ---- Kernel 2: gather q_main + DFT along u=27 (-> 14 bins) ----------------
__global__ void dft_q_kernel(const float* __restrict__ queries,
                             const int* __restrict__ sample_index,
                             float* qf_re, float* qf_im) {
  int id = blockIdx.x * blockDim.x + threadIdx.x;
  if (id >= B_ * NFQ * H_ * D_) return;
  int d = id & 63;
  int h = (id >> 6) & 7;
  int rem = id >> 9;           // b*NFQ + f
  int f = rem % NFQ;
  int b = rem / NFQ;
  float step = -6.2831853071795864f * (float)f / 27.0f;
  float re = 0.f, im = 0.f;
  for (int t = 0; t < U_; ++t) {
    int l = sample_index[t * H_ + h];
    float x = queries[((b * L_ + l) * H_ + h) * D_ + d];
    float sn, cs;
    __sincosf(step * (float)t, &sn, &cs);
    re += x * cs; im += x * sn;
  }
  qf_re[id] = re; qf_im[id] = im;
}

// ---- Kernel 3: complex qk = qf x kf over d (WMMA: 4 real GEMMs) -----------
__global__ __launch_bounds__(32)
void qk_wmma_kernel(const float* __restrict__ qf_re, const float* __restrict__ qf_im,
                    const float* __restrict__ kf_re, const float* __restrict__ kf_im,
                    float* __restrict__ score1) {
  __shared__ _Float16 aR[16 * 32], aI[16 * 32], bR[16 * 32], bI[16 * 32];
  int bx = blockIdx.x;
  int nt = bx % 9;  bx /= 9;       // 9 tiles cover 129 s-bins
  int h  = bx % H_;
  int b  = bx / H_;
  int lane = threadIdx.x;
  v8f rr = {}, ii = {}, ri = {}, ir = {};
  for (int kk = 0; kk < 2; ++kk) {               // d = kk*32 .. kk*32+31
#pragma unroll
    for (int i = 0; i < 16; ++i) {
      int idx = lane * 16 + i;
      int r = idx >> 5, c = idx & 31;
      int d = kk * 32 + c;
      float vr = 0.f, vi = 0.f;                  // A row l = r (pad 14->16)
      if (r < NFQ) {
        int qo = ((b * NFQ + r) * H_ + h) * D_ + d;
        vr = qf_re[qo]; vi = qf_im[qo];
      }
      aR[idx] = (_Float16)vr; aI[idx] = (_Float16)vi;
      int s = nt * 16 + r;                       // B^T row n = s (pad 129->144)
      float wr = 0.f, wi = 0.f;
      if (s < NFK) {
        int ko = ((b * NFK + s) * H_ + h) * D_ + d;
        wr = kf_re[ko]; wi = kf_im[ko];
      }
      bR[idx] = (_Float16)wr; bI[idx] = (_Float16)wi;
    }
    __syncthreads();
    v16h fAR = load_fragA(aR), fAI = load_fragA(aI);
    v16h fBR = load_fragB(bR), fBI = load_fragB(bI);
    rr = __builtin_amdgcn_wmma_f32_16x16x32_f16(false, fAR, false, fBR, (short)0, rr, false, false);
    ii = __builtin_amdgcn_wmma_f32_16x16x32_f16(false, fAI, false, fBI, (short)0, ii, false, false);
    ri = __builtin_amdgcn_wmma_f32_16x16x32_f16(false, fAR, false, fBI, (short)0, ri, false, false);
    ir = __builtin_amdgcn_wmma_f32_16x16x32_f16(false, fAI, false, fBR, (short)0, ir, false, false);
    __syncthreads();
  }
#pragma unroll
  for (int v = 0; v < 8; ++v) {
    int m = v + ((lane >> 4) << 3);
    int n = lane & 15;
    int s = nt * 16 + n;
    if (m < NFQ && s < NFK) {
      float re = rr[v] - ii[v];
      float im = ri[v] + ir[v];
      score1[((b * H_ + h) * NFQ + m) * NFK + s] = sqrtf(re * re + im * im) * 0.125f;
    }
  }
}

// ---- Kernel 4: in-place row softmax (width <= 256) ------------------------
// Stages the row into LDS with CDNA5 async global->LDS DMA (ASYNCcnt-tracked,
// bypasses VGPRs), then reduces.
__global__ void softmax_rows_kernel(float* __restrict__ data, int rows, int width) {
  __shared__ float vals[256];
  __shared__ float red[256];
  int row = blockIdx.x;
  if (row >= rows) return;
  float* p = data + (size_t)row * width;
  int tid = threadIdx.x;
  if (tid < width) {
    unsigned lds_off = (unsigned)(size_t)(&vals[tid]);   // low 32 bits = LDS offset
    const float* gp = p + tid;
    asm volatile("global_load_async_to_lds_b32 %0, %1, off"
                 :: "v"(lds_off), "v"(gp) : "memory");
  }
  asm volatile("s_wait_asynccnt 0x0" ::: "memory");
  __syncthreads();
  float v = (tid < width) ? vals[tid] : -1e30f;
  red[tid] = v;
  __syncthreads();
  for (int s = 128; s > 0; s >>= 1) {
    if (tid < s) red[tid] = fmaxf(red[tid], red[tid + s]);
    __syncthreads();
  }
  float mx = red[0];
  __syncthreads();
  float e = (tid < width) ? __expf(v - mx) : 0.f;
  red[tid] = e;
  __syncthreads();
  for (int s = 128; s > 0; s >>= 1) {
    if (tid < s) red[tid] += red[tid + s];
    __syncthreads();
  }
  float inv = 1.f / red[0];
  if (tid < width) p[tid] = e * inv;
}

// ---- Kernel 5: out_f = score1 x vf (complex, tiny) ------------------------
__global__ void outf_kernel(const float* __restrict__ score1,
                            const float* __restrict__ vf_re, const float* __restrict__ vf_im,
                            float* of_re, float* of_im) {
  int id = blockIdx.x * blockDim.x + threadIdx.x;
  if (id >= B_ * NFQ * H_ * D_) return;
  int d = id & 63;
  int h = (id >> 6) & 7;
  int rem = id >> 9;
  int f = rem % NFQ;
  int b = rem / NFQ;
  const float* srow = score1 + ((size_t)(b * H_ + h) * NFQ + f) * NFK;
  float re = 0.f, im = 0.f;
  for (int s = 0; s < NFK; ++s) {
    float sc = srow[s];
    int o = ((b * NFK + s) * H_ + h) * D_ + d;
    re += sc * vf_re[o];
    im += sc * vf_im[o];
  }
  of_re[id] = re; of_im[id] = im;
}

// ---- Kernel 6: irfft(n=27) + scatter via sample_index ---------------------
__global__ void irfft_scatter_kernel(const float* __restrict__ of_re,
                                     const float* __restrict__ of_im,
                                     const int* __restrict__ sample_index,
                                     float* __restrict__ out) {
  int id = blockIdx.x * blockDim.x + threadIdx.x;
  if (id >= B_ * U_ * H_ * D_) return;
  int d = id & 63;
  int h = (id >> 6) & 7;
  int rem = id >> 9;
  int t = rem % U_;
  int b = rem / U_;
  float acc = of_re[((b * NFQ + 0) * H_ + h) * D_ + d];   // f=0 term (imag ~ 0)
  for (int f = 1; f < NFQ; ++f) {
    float sn, cs;
    __sincosf(6.2831853071795864f * (float)(f * t) / 27.0f, &sn, &cs);
    int o = ((b * NFQ + f) * H_ + h) * D_ + d;
    acc += 2.f * (of_re[o] * cs - of_im[o] * sn);
  }
  acc *= (1.0f / 27.0f);
  int l = sample_index[t * H_ + h];
  out[(size_t)(b * L_ + l) * (H_ * D_) + h * D_ + d] = acc;
}

// ---- Kernel 7: fused time-aware logits + softmax --------------------------
// One 256-thread block per (b,h,l) row: q-row and w_v staged once in LDS,
// logits[s] = sum_d v_tanh(q+k)*w_v computed per thread, then block softmax,
// final score written directly (no second pass over the 3.75MB tensor).
__global__ __launch_bounds__(256)
void time_score_kernel(const float* __restrict__ queries,
                       const float* __restrict__ keys,
                       const float* __restrict__ w_v,
                       const int* __restrict__ rest_index,
                       float* __restrict__ score2) {
  __shared__ float qrow[D_];
  __shared__ float wv[D_];
  __shared__ float red[256];
  int row = blockIdx.x;                 // (b*H + h)*NREST + l
  int l = row % NREST;
  int bh = row / NREST;
  int h = bh & 7;
  int b = bh >> 3;
  int tid = threadIdx.x;
  if (tid < D_) {
    int lq = rest_index[l * H_ + h];
    qrow[tid] = queries[((size_t)(b * L_ + lq) * H_ + h) * D_ + tid];
    wv[tid] = w_v[tid];
  }
  __syncthreads();
  int s = tid;
  const float* kp = keys + ((size_t)(b * S_ + s) * H_ + h) * D_;
  float acc = 0.f;
#pragma unroll 8
  for (int d = 0; d < D_; ++d) acc += htanh(qrow[d] + kp[d]) * wv[d];
  // block softmax over s = 0..255
  red[tid] = acc;
  __syncthreads();
  for (int st = 128; st > 0; st >>= 1) {
    if (tid < st) red[tid] = fmaxf(red[tid], red[tid + st]);
    __syncthreads();
  }
  float mx = red[0];
  __syncthreads();
  float e = __expf(acc - mx);
  red[tid] = e;
  __syncthreads();
  for (int st = 128; st > 0; st >>= 1) {
    if (tid < st) red[tid] += red[tid + st];
    __syncthreads();
  }
  score2[(size_t)row * S_ + s] = e * (1.f / red[0]);
}

// ---- Kernel 8: out_rest = score2 x V (the flagship WMMA GEMM) -------------
__global__ __launch_bounds__(32)
void av_wmma_kernel(const float* __restrict__ score2,
                    const float* __restrict__ values,
                    const int* __restrict__ rest_index,
                    float* __restrict__ out) {
  __shared__ _Float16 aT[16 * 32], bT[16 * 32];
  int bx = blockIdx.x;
  int nt = bx & 3;  bx >>= 2;     // 4 d-tiles
  int lt = bx % 15; bx /= 15;     // 15 l-tiles cover 229
  int h = bx % H_;
  int b = bx / H_;
  int lane = threadIdx.x;
  v8f acc = {};
  for (int kk = 0; kk < 8; ++kk) {
#pragma unroll
    for (int i = 0; i < 16; ++i) {
      int idx = lane * 16 + i;
      int r = idx >> 5, c = idx & 31;
      int s = kk * 32 + c;
      int l = lt * 16 + r;
      float av = 0.f;
      if (l < NREST) av = score2[((size_t)(b * H_ + h) * NREST + l) * S_ + s];
      aT[idx] = (_Float16)av;
      int d = nt * 16 + r;                       // B^T[n=d][k=s]
      bT[idx] = (_Float16)values[((size_t)(b * S_ + s) * H_ + h) * D_ + d];
    }
    __syncthreads();
    v16h fA = load_fragA(aT);
    v16h fB = load_fragB(bT);
    acc = __builtin_amdgcn_wmma_f32_16x16x32_f16(false, fA, false, fB, (short)0, acc, false, false);
    __syncthreads();
  }
#pragma unroll
  for (int v = 0; v < 8; ++v) {
    int m = v + ((lane >> 4) << 3);
    int n = lane & 15;
    int l = lt * 16 + m;
    if (l < NREST) {
      int li = rest_index[l * H_ + h];
      out[(size_t)(b * L_ + li) * (H_ * D_) + h * D_ + (nt * 16 + n)] = acc[v];
    }
  }
}

extern "C" void kernel_launch(void* const* d_in, const int* in_sizes, int n_in,
                              void* d_out, int out_size, void* d_ws, size_t ws_size,
                              hipStream_t stream) {
  const float* queries      = (const float*)d_in[0];
  const float* keys         = (const float*)d_in[1];
  const float* values       = (const float*)d_in[2];
  const float* w_v          = (const float*)d_in[3];
  const int*   sample_index = (const int*)d_in[4];
  const int*   rest_index   = (const int*)d_in[5];

  float* out    = (float*)d_out;
  float* score1 = out + B_ * L_ * H_ * D_;              // (B,H,14,129)
  float* score2 = score1 + B_ * H_ * NFQ * NFK;         // (B,H,229,256)

  const int KV = B_ * NFK * H_ * D_;                    // 132096
  const int QF = B_ * NFQ * H_ * D_;                    // 14336
  float* ws    = (float*)d_ws;
  float* kf_re = ws;
  float* kf_im = kf_re + KV;
  float* vf_re = kf_im + KV;
  float* vf_im = vf_re + KV;
  float* qf_re = vf_im + KV;
  float* qf_im = qf_re + QF;
  float* of_re = qf_im + QF;
  float* of_im = of_re + QF;

  // Frequency branch
  dft_kv_kernel<<<(KV + 255) / 256, 256, 0, stream>>>(keys, values, kf_re, kf_im, vf_re, vf_im);
  dft_q_kernel<<<(QF + 255) / 256, 256, 0, stream>>>(queries, sample_index, qf_re, qf_im);
  qk_wmma_kernel<<<B_ * H_ * 9, 32, 0, stream>>>(qf_re, qf_im, kf_re, kf_im, score1);
  softmax_rows_kernel<<<B_ * H_ * NFQ, 256, 0, stream>>>(score1, B_ * H_ * NFQ, NFK);
  outf_kernel<<<(QF + 255) / 256, 256, 0, stream>>>(score1, vf_re, vf_im, of_re, of_im);
  irfft_scatter_kernel<<<(B_ * U_ * H_ * D_ + 255) / 256, 256, 0, stream>>>(of_re, of_im, sample_index, out);

  // Time branch (fused logits + softmax)
  time_score_kernel<<<B_ * H_ * NREST, 256, 0, stream>>>(queries, keys, w_v, rest_index, score2);
  av_wmma_kernel<<<B_ * H_ * 15 * 4, 32, 0, stream>>>(score2, values, rest_index, out);
}
